// sLSTM_80418967650999
// MI455X (gfx1250) — matrix operand order repaired and checked by reading, not compiled, not run on hardware
//
#include <hip/hip_runtime.h>
#include <hip/hip_bf16.h>
#include <math.h>

// ---------------------------------------------------------------------------
// sLSTM for MI455X (gfx1250): split-bf16 WMMA GEMM with double-buffered
// async global->LDS tile fills, fused gate epilogue (hw tanh/exp), then
// D-parallel time scan with atomic RMS partial sums, then normalize.
// ---------------------------------------------------------------------------

typedef __attribute__((ext_vector_type(16))) __bf16 v16bf;  // 32B, 8 VGPRs
typedef __attribute__((ext_vector_type(8)))  __bf16 v8bf;   // 16B
typedef __attribute__((ext_vector_type(8)))  float  v8f;    // 32B, 8 VGPRs
typedef __attribute__((ext_vector_type(4)))  int    v4i;    // 16B

#define D_DIM 1024
#define N_DIM 3072
#define T_DIM 2048
#define B_DIM 16
#define M_DIM (B_DIM * T_DIM)          // 32768 rows
#define LDSTRIDE 40                    // bf16 elems per LDS row (80B, 16B aligned)

// ---- gfx1250 async global->LDS path (guarded; falls back to sync copies) ---
#if defined(__gfx1250__) && __has_builtin(__builtin_amdgcn_global_load_async_to_lds_b128)
#define HAVE_ASYNC_LDS 1
#else
#define HAVE_ASYNC_LDS 0
#endif

#if HAVE_ASYNC_LDS
#if __has_builtin(__builtin_amdgcn_s_wait_asynccnt)
#define WAIT_ASYNC(n) __builtin_amdgcn_s_wait_asynccnt(n)
#else
#define WAIT_ASYNC(n) asm volatile("s_wait_asynccnt " #n)
#endif
typedef __attribute__((address_space(1))) v4i* gv4i_p;  // global int4*
typedef __attribute__((address_space(3))) v4i* lv4i_p;  // LDS int4*
#else
#define WAIT_ASYNC(n)
#endif

__device__ __forceinline__ void copy16(__bf16* lds, const __bf16* gmem) {
#if HAVE_ASYNC_LDS
    __builtin_amdgcn_global_load_async_to_lds_b128(
        (gv4i_p)(uintptr_t)gmem,
        (lv4i_p)(uintptr_t)lds,
        0, 0);
#else
    *(v8bf*)lds = *(const v8bf*)gmem;
#endif
}

__device__ __forceinline__ float fast_tanh(float x) {
#if __has_builtin(__builtin_amdgcn_tanhf)
    return __builtin_amdgcn_tanhf(x);           // v_tanh_f32
#elif __has_builtin(__builtin_amdgcn_tanh_f32)
    return __builtin_amdgcn_tanh_f32(x);
#else
    return tanhf(x);
#endif
}

__device__ __forceinline__ float softcap15(float v) {
    return 15.0f * fast_tanh(v * (1.0f / 15.0f));
}

// ---------------------------------------------------------------------------
// Kernel 0a: split x (f32) into bf16 hi/lo pair (hi + lo ~= fp32 value)
// ---------------------------------------------------------------------------
__global__ void split_x_kernel(const float* __restrict__ x,
                               __bf16* __restrict__ xh, __bf16* __restrict__ xl,
                               int n) {
    int idx = blockIdx.x * blockDim.x + threadIdx.x;
    if (idx >= n) return;
    float v = x[idx];
    __bf16 h = (__bf16)v;
    xh[idx] = h;
    xl[idx] = (__bf16)(v - (float)h);
}

// ---------------------------------------------------------------------------
// Kernel 0b: split W (f32 [K=1024, N=3072]) into transposed bf16 hi/lo
// WT[n*1024 + k] so B tiles are K-contiguous (same fill path as A tiles).
// ---------------------------------------------------------------------------
__global__ void split_w_kernel(const float* __restrict__ W,
                               __bf16* __restrict__ wth, __bf16* __restrict__ wtl) {
    int idx = blockIdx.x * blockDim.x + threadIdx.x;
    if (idx >= D_DIM * N_DIM) return;
    int k = idx / N_DIM;
    int n = idx % N_DIM;
    float v = W[idx];                       // coalesced read
    __bf16 h = (__bf16)v;
    int t = n * D_DIM + k;
    wth[t] = h;
    wtl[t] = (__bf16)(v - (float)h);
}

__global__ void zero_kernel(float* __restrict__ p, int n) {
    int idx = blockIdx.x * blockDim.x + threadIdx.x;
    if (idx < n) p[idx] = 0.0f;
}

// ---------------------------------------------------------------------------
// LDS fragment loaders (wave32 WMMA layouts, ISA 7.12.2).
// A 16x32 bf16: lane<16 row M=lane, VGPR0-3 = K0..7, VGPR4-7 = K16..23;
//               lane>=16 same rows, K8..15 / K24..31.
// B 32x16 bf16: lane&15 = column N; same K split as A (K0..7/K16..23, K8..15/K24..31).
// LDS rows are K-contiguous with 40-element (80B) stride -> two ds_load_b128.
// ---------------------------------------------------------------------------
__device__ __forceinline__ v16bf ld_fragA(const __bf16* p, int row, int sel) {
    const __bf16* q = p + row * LDSTRIDE + sel * 8;
    v8bf a = *(const v8bf*)(q);
    v8bf b = *(const v8bf*)(q + 16);
    return __builtin_shufflevector(a, b, 0,1,2,3,4,5,6,7,8,9,10,11,12,13,14,15);
}

__device__ __forceinline__ v16bf ld_fragB(const __bf16* p, int col, int sel) {
    const __bf16* q = p + col * LDSTRIDE + sel * 8;
    v8bf a = *(const v8bf*)(q);
    v8bf b = *(const v8bf*)(q + 16);
    return __builtin_shufflevector(a, b, 0,1,2,3,4,5,6,7,8,9,10,11,12,13,14,15);
}

// ---------------------------------------------------------------------------
// Kernel 1: GEMM  pre = x @ W + b  with fused gate activations.
// Block = 256 threads (8 waves, 2x4), block tile 128M x 128N, K-step 32,
// double-buffered LDS stages filled with async-to-LDS (ASYNCcnt pipelined).
// Each wave: 4x2 = 8 16x16 f32 accum tiles; 3 bf16 WMMAs per tile per K-step
// (hi*hi + lo*hi + hi*lo) for fp32-like precision.
// ---------------------------------------------------------------------------
__global__ __launch_bounds__(256)
void gemm_gate_kernel(const __bf16* __restrict__ xh, const __bf16* __restrict__ xl,
                      const __bf16* __restrict__ wth, const __bf16* __restrict__ wtl,
                      const float* __restrict__ bvec, const float* __restrict__ ibias,
                      float* __restrict__ gi, float* __restrict__ gf,
                      float* __restrict__ go) {
    __shared__ __bf16 Ah[2][128 * LDSTRIDE] __attribute__((aligned(16)));
    __shared__ __bf16 Al[2][128 * LDSTRIDE] __attribute__((aligned(16)));
    __shared__ __bf16 Bh[2][128 * LDSTRIDE] __attribute__((aligned(16)));
    __shared__ __bf16 Bl[2][128 * LDSTRIDE] __attribute__((aligned(16)));

    const int tid  = threadIdx.x;
    const int lane = tid & 31;
    const int wv   = tid >> 5;           // 0..7
    const int wm   = wv >> 2;            // 0..1  (64-row slab)
    const int wn   = wv & 3;             // 0..3  (32-col slab)
    const int bm   = blockIdx.x * 128;   // M offset
    const int bn   = blockIdx.y * 128;   // N offset

    const int lsel = (lane >> 4) & 1;    // lane half selects K sub-group
    const int lrow = lane & 15;

    // per-thread copy geometry: 2 segments of 16B per array per stage
    const int cr = tid >> 2;             // row/col 0..63 (+64 for seg 1)
    const int cc = (tid & 3) * 8;        // bf16 offset within K-row

    auto issue_stage = [&](int buf, int k0) {
#pragma unroll
        for (int s = 0; s < 2; ++s) {
            int r = cr + (s << 6);
            size_t ga = (size_t)(bm + r) * D_DIM + k0 + cc;
            size_t gb = (size_t)(bn + r) * D_DIM + k0 + cc;
            int lo = r * LDSTRIDE + cc;
            copy16(&Ah[buf][lo], &xh[ga]);
            copy16(&Al[buf][lo], &xl[ga]);
            copy16(&Bh[buf][lo], &wth[gb]);
            copy16(&Bl[buf][lo], &wtl[gb]);
        }
    };

    v8f acc[4][2];
#pragma unroll
    for (int i = 0; i < 4; ++i)
#pragma unroll
        for (int j = 0; j < 2; ++j) acc[i][j] = (v8f){0,0,0,0,0,0,0,0};

    issue_stage(0, 0);                               // prologue: stage 0

    const int NK = D_DIM / 32;                       // 32 K-steps
    for (int ks = 0; ks < NK; ++ks) {
        const int cur = ks & 1;
        if (ks + 1 < NK) {
            issue_stage(cur ^ 1, (ks + 1) * 32);     // overlap next stage
            WAIT_ASYNC(8);                           // oldest 8 (= cur stage) done
        } else {
            WAIT_ASYNC(0);
        }
        __syncthreads();                             // cur stage visible to all waves

        v16bf bhv[2], blv[2], ahv[4], alv[4];
#pragma unroll
        for (int j = 0; j < 2; ++j) {
            int col = wn * 32 + j * 16 + lrow;
            bhv[j] = ld_fragB(Bh[cur], col, lsel);
            blv[j] = ld_fragB(Bl[cur], col, lsel);
        }
#pragma unroll
        for (int i = 0; i < 4; ++i) {
            int row = wm * 64 + i * 16 + lrow;
            ahv[i] = ld_fragA(Ah[cur], row, lsel);
            alv[i] = ld_fragA(Al[cur], row, lsel);
        }
#pragma unroll
        for (int i = 0; i < 4; ++i) {
#pragma unroll
            for (int j = 0; j < 2; ++j) {
                acc[i][j] = __builtin_amdgcn_wmma_f32_16x16x32_bf16(
                    false, ahv[i], false, bhv[j], (short)0, acc[i][j], false, false);
                acc[i][j] = __builtin_amdgcn_wmma_f32_16x16x32_bf16(
                    false, alv[i], false, bhv[j], (short)0, acc[i][j], false, false);
                acc[i][j] = __builtin_amdgcn_wmma_f32_16x16x32_bf16(
                    false, ahv[i], false, blv[j], (short)0, acc[i][j], false, false);
            }
        }
        __syncthreads();                             // done reading cur before refill
    }

    // Epilogue: C/D layout => lane<16: (M=v, N=lane); lane>=16: (M=v+8, N=lane-16)
    const int gate = bn >> 10;                       // uniform per block
    float* dst = (gate == 0) ? gi : ((gate == 1) ? gf : go);
#pragma unroll
    for (int i = 0; i < 4; ++i) {
#pragma unroll
        for (int j = 0; j < 2; ++j) {
            int col = bn + wn * 32 + j * 16 + lrow;  // 0..3071
            int d   = col & (D_DIM - 1);
            float bb = bvec[col];
            float extra = (gate == 0) ? ibias[d] : 0.0f;
            int row0 = bm + wm * 64 + i * 16 + (lsel << 3);
#pragma unroll
            for (int v = 0; v < 8; ++v) {
                float val = acc[i][j][v] + bb + extra;
                if (gate < 2) val = __expf(softcap15(val));
                dst[(size_t)(row0 + v) * D_DIM + d] = val;
            }
        }
    }
}

// ---------------------------------------------------------------------------
// Kernel 2: time scan. Grid (B=16, D/256=4), 256 threads; each thread owns one
// d-channel of one batch, scans T sequentially, writes un-normalized h to out,
// accumulates block-partial sum(h^2) into sumsq[b*T+t] via atomicAdd.
// ---------------------------------------------------------------------------
__global__ __launch_bounds__(256)
void scan_kernel(const float* __restrict__ x,
                 const float* __restrict__ gi, const float* __restrict__ gf,
                 const float* __restrict__ go,
                 float* __restrict__ out, float* __restrict__ sumsq) {
    const int b   = blockIdx.x;
    const int tid = threadIdx.x;
    const int d   = blockIdx.y * 256 + tid;
    __shared__ float red[8];

    float m = 0.0f;
    size_t base = ((size_t)b * T_DIM) * D_DIM + d;
    for (int t = 0; t < T_DIM; ++t) {
        size_t idx = base + (size_t)t * D_DIM;
        float it = gi[idx];
        float ft = gf[idx];
        float ot = go[idx];
        float xt = x[idx];
        if (t + 1 < T_DIM) {                       // hide next step's latency
            __builtin_prefetch(&gi[idx + D_DIM], 0, 1);
            __builtin_prefetch(&gf[idx + D_DIM], 0, 1);
            __builtin_prefetch(&go[idx + D_DIM], 0, 1);
            __builtin_prefetch(&x[idx + D_DIM], 0, 1);
        }
        m = fmaf(ft, m, it * xt);
        float h = ot * m;
        out[idx] = h;

        float s = h * h;
#pragma unroll
        for (int off = 16; off > 0; off >>= 1) s += __shfl_xor(s, off, 32);
        if ((tid & 31) == 0) red[tid >> 5] = s;
        __syncthreads();
        if (tid == 0) {
            float tot = red[0] + red[1] + red[2] + red[3]
                      + red[4] + red[5] + red[6] + red[7];
            atomicAdd(&sumsq[b * T_DIM + t], tot);
        }
        __syncthreads();
    }
}

// ---------------------------------------------------------------------------
// Kernel 3: normalize in place: out *= scale[d] * rsqrt(mean(h^2) + eps)
// ---------------------------------------------------------------------------
__global__ void norm_kernel(float* __restrict__ out,
                            const float* __restrict__ sumsq,
                            const float* __restrict__ scale, int n) {
    int idx = blockIdx.x * blockDim.x + threadIdx.x;
    if (idx >= n) return;
    int row = idx >> 10;
    int d   = idx & (D_DIM - 1);
    float r = rsqrtf(sumsq[row] * (1.0f / (float)D_DIM) + 1e-8f);
    out[idx] = out[idx] * r * scale[d];
}

// ---------------------------------------------------------------------------
extern "C" void kernel_launch(void* const* d_in, const int* in_sizes, int n_in,
                              void* d_out, int out_size, void* d_ws, size_t ws_size,
                              hipStream_t stream) {
    (void)in_sizes; (void)n_in; (void)out_size; (void)ws_size;
    const float* x   = (const float*)d_in[0];   // [16,2048,1024]
    const float* W   = (const float*)d_in[1];   // [1024,3072]
    const float* bv  = (const float*)d_in[2];   // [3072]
    const float* ib  = (const float*)d_in[3];   // [1024]
    const float* sc  = (const float*)d_in[4];   // [1024]
    float* out = (float*)d_out;                 // [16,2048,1024]

    const size_t MD = (size_t)M_DIM * D_DIM;    // 33,554,432
    const size_t WN = (size_t)D_DIM * N_DIM;    // 3,145,728

    char* p = (char*)d_ws;
    float*  gi   = (float*)p;   p += MD * sizeof(float);
    float*  gf   = (float*)p;   p += MD * sizeof(float);
    float*  go   = (float*)p;   p += MD * sizeof(float);
    __bf16* xh   = (__bf16*)p;  p += MD * sizeof(__bf16);
    __bf16* xl   = (__bf16*)p;  p += MD * sizeof(__bf16);
    __bf16* wth  = (__bf16*)p;  p += WN * sizeof(__bf16);
    __bf16* wtl  = (__bf16*)p;  p += WN * sizeof(__bf16);
    float*  ssq  = (float*)p;   p += (size_t)B_DIM * T_DIM * sizeof(float);

    // 0) precision-split conversions + zero the RMS accumulator
    split_x_kernel<<<(int)((MD + 255) / 256), 256, 0, stream>>>(x, xh, xl, (int)MD);
    split_w_kernel<<<(int)((WN + 255) / 256), 256, 0, stream>>>(W, wth, wtl);
    zero_kernel<<<(B_DIM * T_DIM + 255) / 256, 256, 0, stream>>>(ssq, B_DIM * T_DIM);

    // 1) WMMA GEMM + fused gate activations (async double-buffered tiles)
    dim3 ggrid(M_DIM / 128, N_DIM / 128);       // (256, 24)
    gemm_gate_kernel<<<ggrid, 256, 0, stream>>>(xh, xl, wth, wtl, bv, ib, gi, gf, go);

    // 2) sequential-T scan, D-parallel, atomic sum(h^2)
    dim3 sgrid(B_DIM, D_DIM / 256);             // (16, 4)
    scan_kernel<<<sgrid, 256, 0, stream>>>(x, gi, gf, go, out, ssq);

    // 3) RMS-normalize in place
    norm_kernel<<<(int)((MD + 255) / 256), 256, 0, stream>>>(out, ssq, sc, (int)MD);
}
